// AqSolModel_16647293239458
// MI455X (gfx1250) — matrix operand order repaired and checked by reading, not compiled, not run on hardware
//
#include <hip/hip_runtime.h>
#include <hip/hip_bf16.h>

typedef __attribute__((ext_vector_type(16))) __bf16 v16bf;
typedef __attribute__((ext_vector_type(8)))  __bf16 v8bf;
typedef __attribute__((ext_vector_type(8)))  float  v8f;

#define NN 50000
#define NE 800000
#define ET 850000   /* NE + NN self loops */
#define F0 64
#define HH 128
#define NG 512
#define NEG_SLOPE 0.2f

// ---------------- utility kernels ----------------

__global__ void fill_kernel(float* __restrict__ p, float v, int n) {
    int i = blockIdx.x * 256 + threadIdx.x;
    if (i < n) p[i] = v;
}

__global__ void tobf16_kernel(const float* __restrict__ in, __bf16* __restrict__ out, int n) {
    int i = blockIdx.x * 256 + threadIdx.x;
    if (i < n) out[i] = (__bf16)in[i];
}

// W [fin, 128] row-major fp32 -> Wt [128][fin] bf16 (each output column contiguous over K)
__global__ void wtrans_kernel(const float* __restrict__ W, __bf16* __restrict__ Wt, int fin) {
    int idx = blockIdx.x * 256 + threadIdx.x;
    if (idx >= fin * HH) return;
    int col = idx & (HH - 1);
    int k   = idx >> 7;
    Wt[(size_t)col * fin + k] = (__bf16)W[(size_t)k * HH + col];
}

// ---------------- WMMA GEMM:  C[M,128] = A[M,K](bf16) * B[K,128](bf16, pre-transposed) + bias ----------------

__global__ void __launch_bounds__(256)
gemm_wmma_bf16(const __bf16* __restrict__ A,   // [M x K] row-major bf16
               const __bf16* __restrict__ Bt,  // [128 x K] bf16 (column-major of B)
               const float*  __restrict__ bias,// [128]
               float* __restrict__ C,          // [M x 128] fp32
               int M, int K) {
    int wave  = threadIdx.x >> 5;
    int lane  = threadIdx.x & 31;
    int strip = blockIdx.x * 8 + wave;          // 16-row strip index
    if (strip * 16 >= M) return;                // wave-uniform exit (EXEC stays all-1 for WMMA)
    int half = lane >> 4;                       // 0: lanes 0-15, 1: lanes 16-31
    int mrow = lane & 15;

    v8f acc[8];
#pragma unroll
    for (int t = 0; t < 8; ++t) acc[t] = (v8f){};

    const __bf16* arow = A + (size_t)(strip * 16 + mrow) * K;
    for (int kk = 0; kk < K; kk += 32) {
        // A fragment per documented 16-bit 16x32 layout:
        // half 0 lanes: K = kk+0..7 and kk+16..23 ; half 1 lanes: K = kk+8..15 and kk+24..31
        v8bf a0 = *(const v8bf*)(arow + kk + half * 8);
        v8bf a1 = *(const v8bf*)(arow + kk + half * 8 + 16);
        v16bf afrag = __builtin_shufflevector(a0, a1,
            0, 1, 2, 3, 4, 5, 6, 7, 8, 9, 10, 11, 12, 13, 14, 15);
#pragma unroll
        for (int t = 0; t < 8; ++t) {
            int col = t * 16 + (lane & 15);
            // B fragment: lane holds column `col`, K = kk+half*8+0..7 and kk+16+half*8+0..7
            v8bf b0 = *(const v8bf*)(Bt + (size_t)col * K + kk + half * 8);
            v8bf b1 = *(const v8bf*)(Bt + (size_t)col * K + kk + half * 8 + 16);
            v16bf bfrag = __builtin_shufflevector(b0, b1,
                0, 1, 2, 3, 4, 5, 6, 7, 8, 9, 10, 11, 12, 13, 14, 15);
            acc[t] = __builtin_amdgcn_wmma_f32_16x16x32_bf16(
                false, afrag, false, bfrag, (short)0, acc[t], false, false);
        }
    }
    // C/D layout: VGPR i -> row (half*8 + i), col = lane&15 within tile
#pragma unroll
    for (int t = 0; t < 8; ++t) {
        int col = t * 16 + (lane & 15);
        float bv = bias[col];
#pragma unroll
        for (int i = 0; i < 8; ++i) {
            int row = strip * 16 + half * 8 + i;
            C[(size_t)row * HH + col] = acc[t][i] + bv;
        }
    }
}

// ---------------- edge kernels ----------------

__device__ inline void atomicMaxF32(float* addr, float val) {
    if (val >= 0.0f) atomicMax((int*)addr, __float_as_int(val));
    else             atomicMin((unsigned int*)addr, __float_as_uint(val));
}

__device__ inline void edge_pair(long e, const int* __restrict__ ei, int& s, int& d) {
    if (e < NE) { s = ei[e]; d = ei[NE + e]; }
    else        { s = d = (int)(e - NE); }
}

// one wave per edge: score = att . leaky_relu(xl[src] + xr[dst]); atomic max per dst
__global__ void __launch_bounds__(256)
edge_score_kernel(const float* __restrict__ xl, const float* __restrict__ xr,
                  const float* __restrict__ att, const int* __restrict__ ei,
                  float* __restrict__ score, float* __restrict__ mnode) {
    int wave = threadIdx.x >> 5;
    int lane = threadIdx.x & 31;
    long e = (long)blockIdx.x * 8 + wave;
    if (e >= ET) return;
    int s, d; edge_pair(e, ei, s, d);
    float4 a = *(const float4*)(xl + (size_t)s * HH + lane * 4);
    float4 b = *(const float4*)(xr + (size_t)d * HH + lane * 4);
    float4 w = *(const float4*)(att + lane * 4);
    float p = 0.f, v;
    v = a.x + b.x; v = v > 0.f ? v : NEG_SLOPE * v; p += w.x * v;
    v = a.y + b.y; v = v > 0.f ? v : NEG_SLOPE * v; p += w.y * v;
    v = a.z + b.z; v = v > 0.f ? v : NEG_SLOPE * v; p += w.z * v;
    v = a.w + b.w; v = v > 0.f ? v : NEG_SLOPE * v; p += w.w * v;
#pragma unroll
    for (int off = 16; off > 0; off >>= 1) p += __shfl_xor(p, off, 32);
    if (lane == 0) {
        score[e] = p;
        atomicMaxF32(&mnode[d], p);
    }
}

// one thread per edge: exp(e - max[dst]) -> alpha (in-place over score); atomic sum per dst
__global__ void edge_exp_kernel(const int* __restrict__ ei,
                                float* __restrict__ score,
                                const float* __restrict__ mnode,
                                float* __restrict__ snode) {
    long e = (long)blockIdx.x * 256 + threadIdx.x;
    if (e >= ET) return;
    int s, d; edge_pair(e, ei, s, d); (void)s;
    float ex = __expf(score[e] - mnode[d]);
    score[e] = ex;
    atomicAdd(&snode[d], ex);
}

// one wave per edge: aggr[dst] += (alpha/sum) * xl[src]
__global__ void __launch_bounds__(256)
edge_aggr_kernel(const float* __restrict__ xl, const int* __restrict__ ei,
                 const float* __restrict__ alpha, const float* __restrict__ snode,
                 float* __restrict__ aggr) {
    int wave = threadIdx.x >> 5;
    int lane = threadIdx.x & 31;
    long e = (long)blockIdx.x * 8 + wave;
    if (e >= ET) return;
    int s, d; edge_pair(e, ei, s, d);
    float w = alpha[e] / (snode[d] + 1e-16f);
    float4 v = *(const float4*)(xl + (size_t)s * HH + lane * 4);
    float* dstp = aggr + (size_t)d * HH + lane * 4;
    atomicAdd(dstp + 0, w * v.x);
    atomicAdd(dstp + 1, w * v.y);
    atomicAdd(dstp + 2, w * v.z);
    atomicAdd(dstp + 3, w * v.w);
}

// h = relu(aggr + bias); emit fp32 and bf16 copies
__global__ void finalize_kernel(const float* __restrict__ aggr, const float* __restrict__ bias,
                                float* __restrict__ hf, __bf16* __restrict__ hb) {
    int idx = blockIdx.x * 256 + threadIdx.x;
    if (idx >= NN * HH) return;
    float v = aggr[idx] + bias[idx & (HH - 1)];
    v = v > 0.f ? v : 0.f;
    hf[idx] = v;
    hb[idx] = (__bf16)v;
}

// ---------------- pooling + MLP ----------------

__global__ void pool_kernel(const float* __restrict__ hf, const int* __restrict__ batch,
                            float* __restrict__ gsum, float* __restrict__ cnt) {
    int idx = blockIdx.x * 256 + threadIdx.x;
    if (idx >= NN * HH) return;
    int n = idx >> 7, f = idx & (HH - 1);
    int g = batch[n];
    atomicAdd(&gsum[(size_t)g * HH + f], hf[idx]);
    if (f == 0) atomicAdd(&cnt[g], 1.0f);
}

__global__ void mlp_kernel(const float* __restrict__ gsum, const float* __restrict__ cnt,
                           const float* __restrict__ lin_w, const float* __restrict__ lin_b,
                           const float* __restrict__ out_w, const float* __restrict__ out_b,
                           float* __restrict__ out) {
    int g = blockIdx.x * 256 + threadIdx.x;
    if (g >= NG) return;
    float c = fmaxf(cnt[g], 1.0f);
    float outv = out_b[0];
    for (int j = 0; j < HH / 2; ++j) {
        float acc = lin_b[j];
        for (int k = 0; k < HH; ++k)
            acc += (gsum[(size_t)g * HH + k] / c) * lin_w[(size_t)k * (HH / 2) + j];
        acc = acc > 0.f ? acc : 0.f;
        outv += acc * out_w[j];
    }
    out[g] = outv;
}

// ---------------- host launch ----------------

extern "C" void kernel_launch(void* const* d_in, const int* in_sizes, int n_in,
                              void* d_out, int out_size, void* d_ws, size_t ws_size,
                              hipStream_t stream) {
    (void)in_sizes; (void)n_in; (void)out_size; (void)ws_size;

    const float* x     = (const float*)d_in[0];
    const int*   ei    = (const int*)d_in[1];
    const int*   batch = (const int*)d_in[2];
    const float* Wl[3] = {(const float*)d_in[3],  (const float*)d_in[9],  (const float*)d_in[15]};
    const float* bl[3] = {(const float*)d_in[4],  (const float*)d_in[10], (const float*)d_in[16]};
    const float* Wr[3] = {(const float*)d_in[5],  (const float*)d_in[11], (const float*)d_in[17]};
    const float* br[3] = {(const float*)d_in[6],  (const float*)d_in[12], (const float*)d_in[18]};
    const float* att[3]= {(const float*)d_in[7],  (const float*)d_in[13], (const float*)d_in[19]};
    const float* bb[3] = {(const float*)d_in[8],  (const float*)d_in[14], (const float*)d_in[20]};
    const float* lin_w = (const float*)d_in[21];
    const float* lin_b = (const float*)d_in[22];
    const float* out_w = (const float*)d_in[23];
    const float* out_b = (const float*)d_in[24];

    // carve workspace (256B aligned chunks)
    char* base = (char*)d_ws;
    size_t off = 0;
    auto carve = [&](size_t bytes) -> void* {
        void* p = base + off;
        off += (bytes + 255) & ~(size_t)255;
        return p;
    };
    __bf16* xb0   = (__bf16*)carve((size_t)NN * F0 * 2);
    __bf16* hb    = (__bf16*)carve((size_t)NN * HH * 2);
    float*  xl    = (float*)carve((size_t)NN * HH * 4);
    float*  xr    = (float*)carve((size_t)NN * HH * 4);
    float*  aggr  = (float*)carve((size_t)NN * HH * 4);
    float*  hf    = (float*)carve((size_t)NN * HH * 4);
    float*  score = (float*)carve((size_t)ET * 4);
    float*  mnode = (float*)carve((size_t)NN * 4);
    float*  snode = (float*)carve((size_t)NN * 4);
    __bf16* wlt[3]; __bf16* wrt[3];
    for (int l = 0; l < 3; ++l) {
        wlt[l] = (__bf16*)carve((size_t)HH * HH * 2);
        wrt[l] = (__bf16*)carve((size_t)HH * HH * 2);
    }
    float* gsum = (float*)carve((size_t)NG * HH * 4);
    float* cnt  = (float*)carve((size_t)NG * 4);

    // --- prep: bf16 weights (transposed) and bf16 input features ---
    for (int l = 0; l < 3; ++l) {
        int fin = (l == 0) ? F0 : HH;
        int n = fin * HH;
        wtrans_kernel<<<(n + 255) / 256, 256, 0, stream>>>(Wl[l], wlt[l], fin);
        wtrans_kernel<<<(n + 255) / 256, 256, 0, stream>>>(Wr[l], wrt[l], fin);
    }
    tobf16_kernel<<<((NN * F0) + 255) / 256, 256, 0, stream>>>(x, xb0, NN * F0);

    const __bf16* cur = xb0;
    int K = F0;
    const int strips = NN / 16;                 // 3125 exactly
    const int gemm_blocks = (strips + 7) / 8;   // 8 waves per block
    const int edge_wave_blocks = (ET + 7) / 8;  // one wave per edge

    for (int l = 0; l < 3; ++l) {
        gemm_wmma_bf16<<<gemm_blocks, 256, 0, stream>>>(cur, wlt[l], bl[l], xl, NN, K);
        gemm_wmma_bf16<<<gemm_blocks, 256, 0, stream>>>(cur, wrt[l], br[l], xr, NN, K);

        fill_kernel<<<(NN + 255) / 256, 256, 0, stream>>>(mnode, -__builtin_inff(), NN);
        fill_kernel<<<(NN + 255) / 256, 256, 0, stream>>>(snode, 0.0f, NN);
        fill_kernel<<<((NN * HH) + 255) / 256, 256, 0, stream>>>(aggr, 0.0f, NN * HH);

        edge_score_kernel<<<edge_wave_blocks, 256, 0, stream>>>(xl, xr, att[l], ei, score, mnode);
        edge_exp_kernel<<<(ET + 255) / 256, 256, 0, stream>>>(ei, score, mnode, snode);
        edge_aggr_kernel<<<edge_wave_blocks, 256, 0, stream>>>(xl, ei, score, snode, aggr);

        finalize_kernel<<<((NN * HH) + 255) / 256, 256, 0, stream>>>(aggr, bb[l], hf, hb);
        cur = hb;
        K = HH;
    }

    fill_kernel<<<((NG * HH) + 255) / 256, 256, 0, stream>>>(gsum, 0.0f, NG * HH);
    fill_kernel<<<(NG + 255) / 256, 256, 0, stream>>>(cnt, 0.0f, NG);
    pool_kernel<<<((NN * HH) + 255) / 256, 256, 0, stream>>>(hf, batch, gsum, cnt);
    mlp_kernel<<<(NG + 255) / 256, 256, 0, stream>>>(gsum, cnt, lin_w, lin_b, out_w, out_b,
                                                     (float*)d_out);
}
